// SortOfCLEVRModel_81595788689783
// MI455X (gfx1250) — matrix operand order repaired and checked by reading, not compiled, not run on hardware
//
#include <hip/hip_runtime.h>
#include <hip/hip_bf16.h>

typedef __bf16 v16bf __attribute__((ext_vector_type(16)));
typedef float  v8f   __attribute__((ext_vector_type(8)));
typedef unsigned int v4u __attribute__((ext_vector_type(4)));

#define EPSV 1e-5f

__device__ __forceinline__ unsigned short f2bf(float f) {
  unsigned u = __builtin_bit_cast(unsigned, f);
  unsigned r = u + 0x7FFFu + ((u >> 16) & 1u);
  return (unsigned short)(r >> 16);
}

// ---------------- Conv + BN + ReLU (fp32, direct) ----------------
template <int CIN>
__global__ void conv_bn_relu(const float* __restrict__ in, const float* __restrict__ w,
                             const float* __restrict__ bias,
                             const float* __restrict__ gamma, const float* __restrict__ beta,
                             const float* __restrict__ mean, const float* __restrict__ var,
                             float* __restrict__ out,
                             int Hin, int Win, int Hout, int Wout, int Cout, int pad, int total) {
  int idx = blockIdx.x * blockDim.x + threadIdx.x;
  if (idx >= total) return;
  int ow = idx % Wout;
  int oh = (idx / Wout) % Hout;
  int co = (idx / (Wout * Hout)) % Cout;
  int b  = idx / (Wout * Hout * Cout);

  float acc = 0.0f;
  const float* wbase = w + co * CIN * 9;
  #pragma unroll
  for (int ci = 0; ci < CIN; ++ci) {
    const float* ibase = in + ((size_t)(b * CIN + ci) * Hin) * Win;
    const float* wc = wbase + ci * 9;
    #pragma unroll
    for (int kh = 0; kh < 3; ++kh) {
      int ih = oh * 2 - pad + kh;
      if (ih < 0 || ih >= Hin) continue;
      #pragma unroll
      for (int kw = 0; kw < 3; ++kw) {
        int iw = ow * 2 - pad + kw;
        if (iw < 0 || iw >= Win) continue;
        acc = fmaf(ibase[ih * Win + iw], wc[kh * 3 + kw], acc);
      }
    }
  }
  float s = gamma[co] * rsqrtf(var[co] + EPSV);
  float sh = beta[co] - (mean[co] - bias[co]) * s;
  out[idx] = fmaxf(fmaf(acc, s, sh), 0.0f);
}

// ---------------- objects -> oi, oj  (26 -> 256 projections) ----------------
__global__ void oioj_kernel(const float* __restrict__ x4, const float* __restrict__ W1,
                            float* __restrict__ oi, float* __restrict__ oj) {
  int bn = blockIdx.x;           // b*49 + n
  int b = bn / 49, n = bn % 49;
  __shared__ float obj[26];
  int t = threadIdx.x;
  if (t < 24)       obj[t]  = x4[(size_t)(b * 24 + t) * 49 + n];
  else if (t == 24) obj[24] = -1.0f + (2.0f / 6.0f) * (float)(n / 7);
  else if (t == 25) obj[25] = -1.0f + (2.0f / 6.0f) * (float)(n % 7);
  __syncthreads();
  float so = 0.0f, sj = 0.0f;
  #pragma unroll
  for (int c = 0; c < 26; ++c) {
    float o = obj[c];
    so = fmaf(o, W1[c * 256 + t], so);
    sj = fmaf(o, W1[(26 + c) * 256 + t], sj);
  }
  oi[(size_t)bn * 256 + t] = so;
  oj[(size_t)bn * 256 + t] = sj;
}

// ---------------- qc = question @ W1[52:60] + b1 ----------------
__global__ void qc_kernel(const float* __restrict__ q, const float* __restrict__ W1,
                          const float* __restrict__ b1, float* __restrict__ qc) {
  int b = blockIdx.x, t = threadIdx.x;
  float s = b1[t];
  #pragma unroll
  for (int k = 0; k < 8; ++k) s = fmaf(q[b * 8 + k], W1[(52 + k) * 256 + t], s);
  qc[(size_t)b * 256 + t] = s;
}

// ---------------- pack g2/g3/g4 weights into bf16 B-fragment layout ----------------
__global__ void pack_kernel(const float* __restrict__ W2, const float* __restrict__ W3,
                            const float* __restrict__ W4, unsigned short* __restrict__ pw) {
  int F = blockIdx.x * 256 + threadIdx.x;
  if (F >= 3 * 65536) return;
  int layer = F >> 16;
  int r = F & 65535;
  int e    = r & 15;
  int lane = (r >> 4) & 31;
  int kc   = (r >> 9) & 7;
  int nt   = r >> 12;
  int k = kc * 32 + ((e >> 3) << 4) + ((lane >> 4) << 3) + (e & 7);   // K row
  int n = nt * 16 + (lane & 15);              // N col
  const float* W = (layer == 0) ? W2 : ((layer == 1) ? W3 : W4);
  pw[F] = f2bf(W[k * 256 + n]);
}

// ---------------- fragment helpers ----------------
__device__ __forceinline__ v16bf load_afrag(const unsigned short* myh, int kc, int lane) {
  union { v4u u[2]; v16bf v; } a;
  a.u[0] = *(const v4u*)(myh + ((kc * 2 + 0) * 32 + lane) * 8);
  a.u[1] = *(const v4u*)(myh + ((kc * 2 + 1) * 32 + lane) * 8);
  return a.v;
}
__device__ __forceinline__ v16bf load_bfrag(const unsigned short* __restrict__ pwl,
                                            int nt, int kc, int lane) {
  union { v4u u[2]; v16bf v; } bb;
  const v4u* bp = (const v4u*)(pwl + (((nt * 8 + kc) * 32 + lane) << 4));
  bb.u[0] = bp[0];
  bb.u[1] = bp[1];
  return bb.v;
}

// ---------------- fused relation network: h0 -> 3 WMMA GEMM layers -> sum ----------------
// 128 threads = 4 waves; each wave owns 32 pair-rows as two 16-row A tiles.
// h stored per wave in exact A-fragment order; every B fragment feeds 2 WMMAs.
__global__ void __launch_bounds__(128)
relnet_kernel(const float* __restrict__ oi, const float* __restrict__ oj,
              const float* __restrict__ qc, const unsigned short* __restrict__ pw,
              const float* __restrict__ b2, const float* __restrict__ b3,
              const float* __restrict__ b4, float* __restrict__ relations) {
  // [wave][tile][kc*2+group][lane][within8] : 4*2*16*32*8*2 = 64 KB
  __shared__ __align__(16) unsigned short hbuf[4][2][16][32][8];
  const int tid  = threadIdx.x;
  const int lane = tid & 31;
  const int wv   = tid >> 5;
  const int b    = blockIdx.y;
  const int p0   = blockIdx.x * 128 + wv * 32;   // first pair index of this wave's 32 rows

  unsigned short* myh0 = &hbuf[wv][0][0][0][0];
  unsigned short* myh1 = myh0 + 4096;            // second 16-row tile (8 KB)
  const float* oib = oi + (size_t)b * 49 * 256;
  const float* ojb = oj + (size_t)b * 49 * 256;
  const float* qcb = qc + (size_t)b * 256;

  // ---- build h0 = relu(oi[i] + oj[j] + qc) directly in A-fragment layout ----
  #pragma unroll 2
  for (int t = 0; t < 32; ++t) {
    // per iteration: all lanes share col-group t, lane = row (0..31)
    int r5  = lane;              // row within wave's 32 rows
    int cg  = t;                 // 0..31, col group of 8
    int c0  = cg * 8;
    int p   = p0 + r5;
    bool valid = (p < 2401);
    int pi = valid ? (p / 49) : 0;
    int pj = valid ? (p % 49) : 0;
    const float* oip = oib + pi * 256 + c0;
    const float* ojp = ojb + pj * 256 + c0;
    const float* qcp = qcb + c0;
    union { v4u u; unsigned short h[8]; } outv;
    #pragma unroll
    for (int e = 0; e < 8; ++e) {
      float x = valid ? fmaxf(oip[e] + ojp[e] + qcp[e], 0.0f) : 0.0f;
      outv.h[e] = f2bf(x);
    }
    int kc = cg >> 2, grp = (cg >> 1) & 1, kb = cg & 1;
    int tile = r5 >> 4, row = r5 & 15;
    *(v4u*)(myh0 + tile * 4096 + ((kc * 2 + grp) * 32 + kb * 16 + row) * 8) = outv.u;
  }

  v16bf afragA[8], afragB[8];

  // ---- layers g2, g3: GEMM(32x256 @ 256x256) + bias + ReLU, result back to LDS ----
  #pragma unroll 1
  for (int layer = 0; layer < 2; ++layer) {
    #pragma unroll
    for (int kc = 0; kc < 8; ++kc) {
      afragA[kc] = load_afrag(myh0, kc, lane);
      afragB[kc] = load_afrag(myh1, kc, lane);
    }
    const unsigned short* pwl = pw + layer * 65536;
    const float* bl = (layer == 0) ? b2 : b3;
    #pragma unroll 1
    for (int nt = 0; nt < 16; ++nt) {
      v8f accA = {0.f, 0.f, 0.f, 0.f, 0.f, 0.f, 0.f, 0.f};
      v8f accB = {0.f, 0.f, 0.f, 0.f, 0.f, 0.f, 0.f, 0.f};
      #pragma unroll
      for (int kc = 0; kc < 8; ++kc) {
        v16bf bf = load_bfrag(pwl, nt, kc, lane);
        accA = __builtin_amdgcn_wmma_f32_16x16x32_bf16(false, afragA[kc], false, bf,
                                                       (short)0, accA, false, false);
        accB = __builtin_amdgcn_wmma_f32_16x16x32_bf16(false, afragB[kc], false, bf,
                                                       (short)0, accB, false, false);
      }
      int n  = nt * 16 + (lane & 15);
      float bn = bl[n];
      // scatter D fragments into next layer's A-fragment layout
      int kc2 = nt >> 1, grp = nt & 1;
      int kb = (lane >> 3) & 1;
      int within = lane & 7;
      int off = ((kc2 * 2 + grp) * 32 + kb * 16) * 8 + within;
      unsigned short* dstA = myh0 + off;
      unsigned short* dstB = myh1 + off;
      int mb = (lane >> 4) << 3;
      #pragma unroll
      for (int v = 0; v < 8; ++v) {
        dstA[(mb + v) * 8] = f2bf(fmaxf(accA[v] + bn, 0.0f));
        dstB[(mb + v) * 8] = f2bf(fmaxf(accB[v] + bn, 0.0f));
      }
    }
  }

  // ---- final layer g4: GEMM + bias + ReLU, masked sum over pairs ----
  #pragma unroll
  for (int kc = 0; kc < 8; ++kc) {
    afragA[kc] = load_afrag(myh0, kc, lane);
    afragB[kc] = load_afrag(myh1, kc, lane);
  }
  __syncthreads();                              // all waves done reading hbuf
  float* relPart = (float*)&hbuf[0][0][0][0][0];   // alias dead hbuf as block partial
  relPart[tid] = 0.0f;
  relPart[tid + 128] = 0.0f;
  __syncthreads();

  const unsigned short* pwl = pw + 2 * 65536;
  #pragma unroll 1
  for (int nt = 0; nt < 16; ++nt) {
    v8f accA = {0.f, 0.f, 0.f, 0.f, 0.f, 0.f, 0.f, 0.f};
    v8f accB = {0.f, 0.f, 0.f, 0.f, 0.f, 0.f, 0.f, 0.f};
    #pragma unroll
    for (int kc = 0; kc < 8; ++kc) {
      v16bf bf = load_bfrag(pwl, nt, kc, lane);
      accA = __builtin_amdgcn_wmma_f32_16x16x32_bf16(false, afragA[kc], false, bf,
                                                     (short)0, accA, false, false);
      accB = __builtin_amdgcn_wmma_f32_16x16x32_bf16(false, afragB[kc], false, bf,
                                                     (short)0, accB, false, false);
    }
    int n  = nt * 16 + (lane & 15);
    float bn = b4[n];
    int mb = (lane >> 4) << 3;
    float psum = 0.0f;
    #pragma unroll
    for (int v = 0; v < 8; ++v) {
      float xA = fmaxf(accA[v] + bn, 0.0f);
      float xB = fmaxf(accB[v] + bn, 0.0f);
      psum += ((p0 + mb + v) < 2401)      ? xA : 0.0f;
      psum += ((p0 + 16 + mb + v) < 2401) ? xB : 0.0f;
    }
    psum += __shfl_xor(psum, 16, 32);
    if (lane < 16) atomicAdd(&relPart[n], psum);
  }
  __syncthreads();
  atomicAdd(&relations[(size_t)b * 256 + tid], relPart[tid]);
  atomicAdd(&relations[(size_t)b * 256 + tid + 128], relPart[tid + 128]);
}

// ---------------- f layers ----------------
__global__ void f1_kernel(const float* __restrict__ rel, const float* __restrict__ Wf,
                          const float* __restrict__ bf, float* __restrict__ r) {
  int b = blockIdx.x, t = threadIdx.x;
  __shared__ float row[256];
  row[t] = rel[(size_t)b * 256 + t];
  __syncthreads();
  float s = bf[t];
  for (int k = 0; k < 256; ++k) s = fmaf(row[k], Wf[k * 256 + t], s);
  r[(size_t)b * 256 + t] = fmaxf(s, 0.0f);
}

__global__ void f2_kernel(const float* __restrict__ r, const float* __restrict__ Wo,
                          const float* __restrict__ bo, float* __restrict__ out) {
  int b = blockIdx.x, lane = threadIdx.x;   // 32 threads
  float s0 = 0.0f, s1 = 0.0f;
  for (int k = lane; k < 256; k += 32) {
    float rv = r[(size_t)b * 256 + k];
    s0 = fmaf(rv, Wo[k * 2 + 0], s0);
    s1 = fmaf(rv, Wo[k * 2 + 1], s1);
  }
  for (int off = 16; off > 0; off >>= 1) {
    s0 += __shfl_down(s0, off, 32);
    s1 += __shfl_down(s1, off, 32);
  }
  if (lane == 0) { out[b * 2 + 0] = s0 + bo[0]; out[b * 2 + 1] = s1 + bo[1]; }
}

// ---------------- launch ----------------
extern "C" void kernel_launch(void* const* d_in, const int* in_sizes, int n_in,
                              void* d_out, int out_size, void* d_ws, size_t ws_size,
                              hipStream_t stream) {
  const float* image    = (const float*)d_in[0];
  const float* question = (const float*)d_in[1];
  // conv_params: (w,b) x4 at 2..9 ; bn_params: (g,be,m,v) x4 at 10..25
  const float* cw[4] = { (const float*)d_in[2], (const float*)d_in[4],
                         (const float*)d_in[6], (const float*)d_in[8] };
  const float* cb[4] = { (const float*)d_in[3], (const float*)d_in[5],
                         (const float*)d_in[7], (const float*)d_in[9] };
  const float* bng[4], *bnb[4], *bnm[4], *bnv[4];
  for (int i = 0; i < 4; ++i) {
    bng[i] = (const float*)d_in[10 + 4 * i + 0];
    bnb[i] = (const float*)d_in[10 + 4 * i + 1];
    bnm[i] = (const float*)d_in[10 + 4 * i + 2];
    bnv[i] = (const float*)d_in[10 + 4 * i + 3];
  }
  const float* W1 = (const float*)d_in[26];
  const float* b1 = (const float*)d_in[27];
  const float* W2 = (const float*)d_in[28];
  const float* b2 = (const float*)d_in[29];
  const float* W3 = (const float*)d_in[30];
  const float* b3 = (const float*)d_in[31];
  const float* W4 = (const float*)d_in[32];
  const float* b4 = (const float*)d_in[33];
  const float* Wf = (const float*)d_in[34];
  const float* bff = (const float*)d_in[35];
  const float* Wo = (const float*)d_in[36];
  const float* bo = (const float*)d_in[37];

  char* ws = (char*)d_ws;
  float* x1 = (float*)(ws + 0);                         // 256*24*64*64 f32 = 100663296 B
  float* x2 = (float*)(ws + 100663296);                 // 256*24*32*32 f32 = 25165824 B
  float* x3 = (float*)(ws + 0);                         // reuse: 256*24*16*16 f32
  float* x4 = (float*)(ws + 6291456);                   // 256*24*7*7 f32
  float* oi = (float*)(ws + 8388608);                   // 256*49*256 f32
  float* oj = (float*)(ws + 21233664);                  // 256*49*256 f32
  float* qc = (float*)(ws + 34078720);                  // 256*256 f32
  unsigned short* pw = (unsigned short*)(ws + 34340864);// 3*65536 bf16
  float* relations = (float*)(ws + 34734080);           // 256*256 f32
  float* rbuf = (float*)(ws + 34996224);                // 256*256 f32

  // conv stack
  {
    int total = 256 * 24 * 64 * 64;
    conv_bn_relu<3><<<(total + 255) / 256, 256, 0, stream>>>(
        image, cw[0], cb[0], bng[0], bnb[0], bnm[0], bnv[0], x1,
        128, 128, 64, 64, 24, 1, total);
  }
  {
    int total = 256 * 24 * 32 * 32;
    conv_bn_relu<24><<<(total + 255) / 256, 256, 0, stream>>>(
        x1, cw[1], cb[1], bng[1], bnb[1], bnm[1], bnv[1], x2,
        64, 64, 32, 32, 24, 1, total);
  }
  {
    int total = 256 * 24 * 16 * 16;
    conv_bn_relu<24><<<(total + 255) / 256, 256, 0, stream>>>(
        x2, cw[2], cb[2], bng[2], bnb[2], bnm[2], bnv[2], x3,
        32, 32, 16, 16, 24, 1, total);
  }
  {
    int total = 256 * 24 * 7 * 7;
    conv_bn_relu<24><<<(total + 255) / 256, 256, 0, stream>>>(
        x3, cw[3], cb[3], bng[3], bnb[3], bnm[3], bnv[3], x4,
        16, 16, 7, 7, 24, 0, total);
  }

  // weight packing
  pack_kernel<<<(3 * 65536 + 255) / 256, 256, 0, stream>>>(W2, W3, W4, pw);

  // oi / oj / qc projections
  oioj_kernel<<<256 * 49, 256, 0, stream>>>(x4, W1, oi, oj);
  qc_kernel<<<256, 256, 0, stream>>>(question, W1, b1, qc);

  // zero relations accumulator (capture-safe)
  hipMemsetAsync(relations, 0, 256 * 256 * sizeof(float), stream);

  // fused relation network: the ~242 GFLOP of WMMA work
  dim3 grid(19, 256);   // 19 tiles of 128 pairs cover 49*49=2401; y = batch
  relnet_kernel<<<grid, 128, 0, stream>>>(oi, oj, qc, pw, b2, b3, b4, relations);

  // f layers
  f1_kernel<<<256, 256, 0, stream>>>(relations, Wf, bff, rbuf);
  f2_kernel<<<256, 32, 0, stream>>>(rbuf, Wo, bo, (float*)d_out);
}
